// MyLSTM_10737418240147
// MI455X (gfx1250) — matrix operand order repaired and checked by reading, not compiled, not run on hardware
//
#include <hip/hip_runtime.h>

// Shapes from the reference
#define BB 512
#define TT 32
#define INP 512
#define CELLN 128
#define ROWS_TOTAL (BB * TT)   // 16384 independent rows
#define ROWS_WG 64             // rows per workgroup
#define NWAVES 4               // 128 threads, wave32
#define NT 8                   // 128 cols / 16
#define KB_X 16                // 512 / 32 k-steps for x part
#define KB_C 4                 // 128 / 32 k-steps for c part

typedef __attribute__((ext_vector_type(16))) __bf16 v16bf;
typedef __attribute__((ext_vector_type(8)))  float  v8f;

static __device__ __forceinline__ v8f wmma_bf16(v16bf a, v16bf b, v8f c) {
  // D = A(16x32 bf16) x B(32x16 bf16) + C(16x16 f32)
  return __builtin_amdgcn_wmma_f32_16x16x32_bf16(
      /*neg_a=*/false, a, /*neg_b=*/false, b,
      /*c_mod=*/(short)0, c, /*reuse_a=*/false, /*reuse_b=*/false);
}

static __device__ __forceinline__ float sigm(float x) {
  return 1.0f / (1.0f + __expf(-x));
}

// A-fragment gather (this code's contiguous k order, same as B frags): lane = row (lane&15),
// K set = sel + {0..15}, sel = (lane>=16)?16:0.
// p must point at (row_base + sel); 32-byte aligned.
static __device__ __forceinline__ v16bf load_a_f32(const float* p) {
  const float4* p4 = (const float4*)p;
  float tmp[16];
  *(float4*)&tmp[0]  = p4[0];
  *(float4*)&tmp[4]  = p4[1];
  *(float4*)&tmp[8]  = p4[2];
  *(float4*)&tmp[12] = p4[3];
  v16bf a;
#pragma unroll
  for (int e = 0; e < 16; ++e) a[e] = (__bf16)tmp[e];
  return a;
}

// B-fragment gather from row-major global W (ld = CELLN):
// lane = col (lane&15), K = klo + e, klo folded into rowbase by caller.
static __device__ __forceinline__ v16bf load_b_global(const float* W, int rowbase, int col) {
  v16bf b;
#pragma unroll
  for (int e = 0; e < 16; ++e) b[e] = (__bf16)W[(rowbase + e) * CELLN + col];
  return b;
}

// pre_g = G[g] + c @ Wrec[g]  (C operand seeds the accumulator)
static __device__ __forceinline__ v8f gate_pre(int g, int nt, const v16bf* cA,
                                               const __bf16* wrec, const float* Glds,
                                               int wave, int lane) {
  v8f acc = *(const v8f*)&Glds[((g * NWAVES + wave) * NT + nt) * 256 + lane * 8];
#pragma unroll
  for (int kb = 0; kb < KB_C; ++kb) {
    v16bf b = *(const v16bf*)&wrec[((g * KB_C + kb) * NT + nt) * 512 + lane * 16];
    acc = wmma_bf16(cA[kb], b, acc);
  }
  return acc;
}

__global__ __launch_bounds__(128) void lstm_fused(
    const float* __restrict__ x,
    const float* __restrict__ Wf,  const float* __restrict__ bf_,
    const float* __restrict__ Wi1, const float* __restrict__ bi1,
    const float* __restrict__ Wi2, const float* __restrict__ bi2,
    const float* __restrict__ Wo,  const float* __restrict__ bo,
    const float* __restrict__ Wc,  const float* __restrict__ bc,
    float* __restrict__ outp, float* __restrict__ coutp) {
  extern __shared__ char smem[];
  __bf16* wrec = (__bf16*)smem;                  // 4*128*128 bf16 = 128 KB (B-frag order)
  float*  Glds = (float*)(smem + 131072);        // 4*4*8*256 f32  = 128 KB (C-frag order)
  float*  ctile = (float*)(smem + 262144);       // 4 waves * 16x128 f32 = 32 KB

  const int tid  = threadIdx.x;
  const int lane = tid & 31;
  const int wave = tid >> 5;
  const int lcol = lane & 15;
  const int hi   = lane >> 4;
  const int selA = hi * 16;   // A-frag K offset
  const int kloB = hi * 16;   // B-frag K offset
  const int hi8  = hi * 8;    // C/D-frag row offset

  const float* Wg[4] = {Wf, Wi1, Wi2, Wo};
  const float* bg[4] = {bf_, bi1, bi2, bo};

  // ---- cooperative fill: recurrent weights (W rows 0..127 = c part) as bf16 B-frags ----
  for (int i = tid; i < 65536; i += 128) {
    int e  = i & 15;
    int ln = (i >> 4) & 31;
    int nt = (i >> 9) & 7;
    int kb = (i >> 12) & 3;
    int g  = (i >> 14) & 3;
    int row = kb * 32 + ((ln >= 16) ? 16 : 0) + e;
    int col = nt * 16 + (ln & 15);
    wrec[i] = (__bf16)Wg[g][row * CELLN + col];
  }
  __syncthreads();

  const int rowBase = blockIdx.x * ROWS_WG + wave * 16;
  const int grow    = rowBase + lcol;  // this lane's A-fragment row

  // ---- Phase A: G[g] = x @ W[g][CELL:,:] + b[g], loop-invariant across scan steps ----
  for (int g = 0; g < 4; ++g) {
    v8f acc[NT];
#pragma unroll
    for (int nt = 0; nt < NT; ++nt) {
      float bv = bg[g][nt * 16 + lcol];
#pragma unroll
      for (int j = 0; j < 8; ++j) acc[nt][j] = bv;
    }
#pragma unroll 1
    for (int kb = 0; kb < KB_X; ++kb) {
      v16bf a = load_a_f32(x + grow * INP + kb * 32 + selA);
#pragma unroll
      for (int nt = 0; nt < NT; ++nt) {
        v16bf b = load_b_global(Wg[g], CELLN + kb * 32 + kloB, nt * 16 + lcol);
        acc[nt] = wmma_bf16(a, b, acc[nt]);
      }
    }
#pragma unroll
    for (int nt = 0; nt < NT; ++nt)
      *(v8f*)&Glds[((g * NWAVES + wave) * NT + nt) * 256 + lane * 8] = acc[nt];
  }
  // No barrier: each wave reads only its own G / ctile regions from here on.

  // ---- Phase B: recurrence, c kept in D-layout accumulator registers ----
  v8f cD[NT];
#pragma unroll
  for (int nt = 0; nt < NT; ++nt) {
#pragma unroll
    for (int j = 0; j < 8; ++j) cD[nt][j] = 0.0f;
  }

  float* myct = ctile + wave * 2048;  // wave-private 16x128 f32 transpose tile
  v16bf cA[KB_C];

#pragma unroll 1
  for (int t = 0; t < TT - 1; ++t) {
    // scatter c (lane=col) -> LDS row-major
#pragma unroll
    for (int nt = 0; nt < NT; ++nt) {
#pragma unroll
      for (int v = 0; v < 8; ++v)
        myct[(v + hi8) * CELLN + nt * 16 + lcol] = cD[nt][v];
    }
    // gather c as A fragments (lane=row), bf16
#pragma unroll
    for (int kb = 0; kb < KB_C; ++kb)
      cA[kb] = load_a_f32(myct + lcol * CELLN + kb * 32 + selA);
    // gates + cell update (o gate is dead except at the last step)
#pragma unroll
    for (int nt = 0; nt < NT; ++nt) {
      v8f pf  = gate_pre(0, nt, cA, wrec, Glds, wave, lane);
      v8f pi1 = gate_pre(1, nt, cA, wrec, Glds, wave, lane);
      v8f pi2 = gate_pre(2, nt, cA, wrec, Glds, wave, lane);
#pragma unroll
      for (int j = 0; j < 8; ++j)
        cD[nt][j] = cD[nt][j] * sigm(pf[j]) + sigm(pi1[j]) * tanhf(pi2[j]);
    }
  }

  // ---- peeled final step: also o gate (vs OLD c), final matmul, softmax ----
#pragma unroll
  for (int nt = 0; nt < NT; ++nt) {
#pragma unroll
    for (int v = 0; v < 8; ++v)
      myct[(v + hi8) * CELLN + nt * 16 + lcol] = cD[nt][v];
  }
#pragma unroll
  for (int kb = 0; kb < KB_C; ++kb)
    cA[kb] = load_a_f32(myct + lcol * CELLN + kb * 32 + selA);

  v8f po[NT];
#pragma unroll
  for (int nt = 0; nt < NT; ++nt)
    po[nt] = gate_pre(3, nt, cA, wrec, Glds, wave, lane);  // uses old c

#pragma unroll
  for (int nt = 0; nt < NT; ++nt) {
    v8f pf  = gate_pre(0, nt, cA, wrec, Glds, wave, lane);
    v8f pi1 = gate_pre(1, nt, cA, wrec, Glds, wave, lane);
    v8f pi2 = gate_pre(2, nt, cA, wrec, Glds, wave, lane);
#pragma unroll
    for (int j = 0; j < 8; ++j)
      cD[nt][j] = cD[nt][j] * sigm(pf[j]) + sigm(pi1[j]) * tanhf(pi2[j]);
  }

  // o = sigmoid(pre_o) * tanh(c_new); store final c to global
  v8f oa[NT];
#pragma unroll
  for (int nt = 0; nt < NT; ++nt) {
#pragma unroll
    for (int j = 0; j < 8; ++j)
      oa[nt][j] = sigm(po[nt][j]) * tanhf(cD[nt][j]);
#pragma unroll
    for (int v = 0; v < 8; ++v)
      coutp[(rowBase + v + hi8) * CELLN + nt * 16 + lcol] = cD[nt][v];
  }

  // logits = oa @ Wc + bc  (transpose oa via the same LDS tile; B frags from global)
#pragma unroll
  for (int nt = 0; nt < NT; ++nt) {
#pragma unroll
    for (int v = 0; v < 8; ++v)
      myct[(v + hi8) * CELLN + nt * 16 + lcol] = oa[nt][v];
  }
  v16bf oA[KB_C];
#pragma unroll
  for (int kb = 0; kb < KB_C; ++kb)
    oA[kb] = load_a_f32(myct + lcol * CELLN + kb * 32 + selA);

  float* lg = Glds + wave * 2048;  // reuse G_f block (wave-private, G is dead now)
#pragma unroll
  for (int nt = 0; nt < NT; ++nt) {
    v8f acc;
    float bv = bc[nt * 16 + lcol];
#pragma unroll
    for (int j = 0; j < 8; ++j) acc[j] = bv;
#pragma unroll
    for (int kb = 0; kb < KB_C; ++kb) {
      v16bf b = load_b_global(Wc, kb * 32 + kloB, nt * 16 + lcol);
      acc = wmma_bf16(oA[kb], b, acc);
    }
#pragma unroll
    for (int v = 0; v < 8; ++v)
      lg[(v + hi8) * CELLN + nt * 16 + lcol] = acc[v];
  }
  __syncthreads();

  // row softmax over 128, lanes 0..15 own one row each
  if (lane < 16) {
    const float* Lr = lg + lcol * CELLN;
    float m = -3.402823466e38f;
    for (int j = 0; j < CELLN; ++j) m = fmaxf(m, Lr[j]);
    float s = 0.0f;
    for (int j = 0; j < CELLN; ++j) s += __expf(Lr[j] - m);
    float inv = 1.0f / s;
    const int r = rowBase + lcol;
    for (int j = 0; j < CELLN; ++j) outp[r * CELLN + j] = __expf(Lr[j] - m) * inv;
  }
}

extern "C" void kernel_launch(void* const* d_in, const int* in_sizes, int n_in,
                              void* d_out, int out_size, void* d_ws, size_t ws_size,
                              hipStream_t stream) {
  (void)in_sizes; (void)n_in; (void)d_ws; (void)ws_size; (void)out_size;
  const float* x   = (const float*)d_in[0];
  const float* Wf  = (const float*)d_in[1];
  const float* bf_ = (const float*)d_in[2];
  const float* Wi1 = (const float*)d_in[3];
  const float* bi1 = (const float*)d_in[4];
  const float* Wi2 = (const float*)d_in[5];
  const float* bi2 = (const float*)d_in[6];
  const float* Wo  = (const float*)d_in[7];
  const float* bo  = (const float*)d_in[8];
  const float* Wc  = (const float*)d_in[9];
  const float* bc  = (const float*)d_in[10];

  float* outp  = (float*)d_out;                     // softmax output [16384,128]
  float* coutp = outp + (size_t)ROWS_TOTAL * CELLN; // final cell state [16384,128]

  dim3 grid(ROWS_TOTAL / ROWS_WG);  // 256 workgroups
  dim3 block(128);                  // 4 waves (wave32)
  size_t lds = 131072 + 131072 + 32768;  // 288 KB <= 320 KB/WGP
  lstm_fused<<<grid, block, lds, stream>>>(x, Wf, bf_, Wi1, bi1, Wi2, bi2,
                                           Wo, bo, Wc, bc, outp, coutp);
}
